// HydraBlock_2396591751257
// MI455X (gfx1250) — matrix operand order repaired and checked by reading, not compiled, not run on hardware
//
#include <hip/hip_runtime.h>

typedef __attribute__((ext_vector_type(16))) _Float16 v16h;
typedef __attribute__((ext_vector_type(8)))  _Float16 v8h;
typedef __attribute__((ext_vector_type(8)))  float    v8f;

#define NTOK 2048   // B*T
#define DMODEL 128
#define NH 8
#define HD 16
#define NEXP 256
#define RLORA 32
#define TSEQ 512
#define EPSF 1e-6f

#define ACT_NONE 0
#define ACT_ELU_ALL 1
#define ACT_ELU_KHALF 2

__device__ __forceinline__ float elu1(float x) {
  const float e = __expf(fminf(x, 0.f));
  return x > 0.f ? x + 1.f : e;            // branchless -> v_cndmask
}
__device__ __forceinline__ float sigf(float x) { return 1.f / (1.f + __expf(-x)); }

// ---------------------------------------------------------------------------
// Weight prep: W[K x M] fp32 -> Wt[M x K] f16 (contiguous B fragments)
// ---------------------------------------------------------------------------
__global__ void wt_prep_kernel(const float* __restrict__ W, _Float16* __restrict__ Wt,
                               int K, int M) {
  const int i = blockIdx.x * 256 + threadIdx.x;
  if (i >= K * M) return;
  const int k = i / M, m = i % M;
  Wt[(size_t)m * K + k] = (_Float16)W[i];
}

// expW[e][d][o] fp32 -> expw16[e][o][d] f16
__global__ void expw_prep_kernel(const float* __restrict__ expW,
                                 _Float16* __restrict__ Wt) {
  const size_t i = (size_t)blockIdx.x * 256 + threadIdx.x;
  if (i >= (size_t)NEXP * DMODEL * DMODEL) return;
  const int e = (int)(i >> 14);
  const int rem = (int)(i & 16383);
  const int d = rem >> 7, o = rem & 127;
  Wt[((size_t)e << 14) + (size_t)o * DMODEL + d] = (_Float16)expW[i];
}

// ---------------------------------------------------------------------------
// GEMM: out[N x M] = act(A[N x K] @ W[K x M] + bias), f16 WMMA / f32 accumulate.
// ---------------------------------------------------------------------------
template <int K, int ACT, bool HAS_BIAS>
__global__ void gemm_wmma(const float* __restrict__ A, const _Float16* __restrict__ Wt,
                          const float* __restrict__ bias, float* __restrict__ out,
                          int M) {
  __shared__ __align__(32) _Float16 As[16 * K];
  const int tid = threadIdx.x;
  const int lane = tid & 31;
  const int wave = tid >> 5;
  const int row0 = blockIdx.x << 4;
  const int col0 = (blockIdx.y * 8 + wave) << 4;

#pragma unroll
  for (int i = tid; i < 16 * K; i += 256) {
    const int r = i / K, k = i % K;
    As[i] = (_Float16)A[(size_t)(row0 + r) * K + k];
  }
  __syncthreads();

  if (col0 < M) {
    const int m    = lane & 15;
    const int half = lane >> 4;
    const _Float16* wcol = Wt + (size_t)(col0 + m) * K;

    v8f acc = {0.f, 0.f, 0.f, 0.f, 0.f, 0.f, 0.f, 0.f};
#pragma unroll
    for (int k0 = 0; k0 < K; k0 += 32) {
      const v8h lo = *(const v8h*)(&As[m * K + k0 + 16 * half]);
      const v8h hi = *(const v8h*)(&As[m * K + k0 + 16 * half + 8]);
      v16h a;
#pragma unroll
      for (int i = 0; i < 8; ++i) { a[i] = lo[i]; a[8 + i] = hi[i]; }
      const v16h b = *(const v16h*)(wcol + k0 + 16 * half);
      acc = __builtin_amdgcn_wmma_f32_16x16x32_f16(false, a, false, b, (short)0, acc,
                                                   false, false);
    }

    const bool do_elu =
        (ACT == ACT_ELU_ALL) || (ACT == ACT_ELU_KHALF && ((col0 & 16) == 0));
    const int col = col0 + m;
    const float bv = HAS_BIAS ? bias[col] : 0.f;
#pragma unroll
    for (int r = 0; r < 8; ++r) {
      const int row = row0 + r + 8 * half;
      float val = acc[r] + bv;
      if (do_elu) val = elu1(val);
      out[(size_t)row * M + col] = val;
    }
  }
}

// ---------------------------------------------------------------------------
__global__ void rmsnorm_kernel(const float* __restrict__ x, const float* __restrict__ g,
                               float* __restrict__ out) {
  const int row = blockIdx.x, tid = threadIdx.x;
  __shared__ float red[DMODEL];
  const float v = x[(size_t)row * DMODEL + tid];
  red[tid] = v * v;
  __syncthreads();
  for (int s = 64; s > 0; s >>= 1) {
    if (tid < s) red[tid] += red[tid + s];
    __syncthreads();
  }
  const float inv = 1.f / sqrtf(red[0] / (float)DMODEL + EPSF);
  out[(size_t)row * DMODEL + tid] = g[tid] * v * inv;
}

// ---------------------------------------------------------------------------
// Linear attention per (b,h).  grid = 32, block = 256 (8 waves).
// ---------------------------------------------------------------------------
__global__ void attn_kernel(const float* __restrict__ q, const float* __restrict__ kv,
                            float* __restrict__ attn) {
  const int b = blockIdx.x / NH, h = blockIdx.x % NH;
  const int tid = threadIdx.x, wave = tid >> 5, lane = tid & 31;
  const int m = lane & 15, half = lane >> 4;
  __shared__ float ctx[8 * 256];
  __shared__ float ksum[HD];
  __shared__ float zbuf[TSEQ];

  v8f acc = {0.f, 0.f, 0.f, 0.f, 0.f, 0.f, 0.f, 0.f};
#pragma unroll
  for (int step = 0; step < 2; ++step) {
    const int t0 = wave * 64 + step * 32;
    v16h a, bf;
#pragma unroll
    for (int i = 0; i < 16; ++i) {
      const int v = i >> 1, w = i & 1;
      const int kla = (v < 4) ? (2 * v + 8 * half + w) : (16 + 2 * (v - 4) + 8 * half + w);
      a[i]  = (_Float16)kv[((size_t)(b * TSEQ + t0 + kla)) * 256 + h * 32 + m];
      const int klb = (v < 4) ? (2 * v + 8 * half + w) : (16 + 2 * (v - 4) + 8 * half + w);
      bf[i] = (_Float16)kv[((size_t)(b * TSEQ + t0 + klb)) * 256 + h * 32 + 16 + m];
    }
    acc = __builtin_amdgcn_wmma_f32_16x16x32_f16(false, a, false, bf, (short)0, acc,
                                                 false, false);
  }
#pragma unroll
  for (int r = 0; r < 8; ++r) ctx[wave * 256 + (r + 8 * half) * 16 + m] = acc[r];
  if (tid < HD) ksum[tid] = 0.f;
  __syncthreads();
  {
    const int d = tid & 15, tg = tid >> 4;
    float s = 0.f;
    for (int t = tg; t < TSEQ; t += 16)
      s += kv[((size_t)(b * TSEQ + t)) * 256 + h * 32 + d];
    atomicAdd(&ksum[d], s);
  }
  __syncthreads();
  {
    float s = 0.f;
#pragma unroll
    for (int w = 0; w < 8; ++w) s += ctx[w * 256 + tid];
    ctx[tid] = s;
  }
  for (int t = tid; t < TSEQ; t += 256) {
    float s = 0.f;
#pragma unroll
    for (int d = 0; d < HD; ++d)
      s += q[((size_t)(b * TSEQ + t)) * DMODEL + h * 16 + d] * ksum[d];
    zbuf[t] = 1.f / (s + EPSF);
  }
  __syncthreads();

#pragma unroll
  for (int j = 0; j < 4; ++j) {
    const int trow0 = (wave * 4 + j) * 16;
    v16h a, bf;
#pragma unroll
    for (int i = 0; i < 16; ++i) {
      const int v = i >> 1, w = i & 1;
      const int kla = (v < 4) ? (2 * v + 8 * half + w) : (16 + 2 * (v - 4) + 8 * half + w);
      a[i] = (kla < 16)
                 ? (_Float16)q[((size_t)(b * TSEQ + trow0 + m)) * DMODEL + h * 16 + kla]
                 : (_Float16)0.f;
      const int klb = (v < 4) ? (2 * v + 8 * half + w) : (16 + 2 * (v - 4) + 8 * half + w);
      bf[i] = (klb < 16) ? (_Float16)ctx[klb * 16 + m] : (_Float16)0.f;
    }
    v8f o = {0.f, 0.f, 0.f, 0.f, 0.f, 0.f, 0.f, 0.f};
    o = __builtin_amdgcn_wmma_f32_16x16x32_f16(false, a, false, bf, (short)0, o,
                                               false, false);
#pragma unroll
    for (int r = 0; r < 8; ++r) {
      const int t = trow0 + r + 8 * half;
      attn[((size_t)(b * TSEQ + t)) * DMODEL + h * 16 + m] = o[r] * zbuf[t];
    }
  }
}

// ---------------------------------------------------------------------------
__global__ void tquant_kernel(const float* __restrict__ y, const float* __restrict__ scale,
                              float* __restrict__ out) {
  const int row = blockIdx.x, tid = threadIdx.x;
  __shared__ float red[DMODEL];
  const float v = y[(size_t)row * DMODEL + tid];
  red[tid] = v * v;
  __syncthreads();
  for (int s = 64; s > 0; s >>= 1) {
    if (tid < s) red[tid] += red[tid + s];
    __syncthreads();
  }
  const float mag = sqrtf(red[0] / (float)DMODEL + EPSF);
  float ph = v / mag;
  ph = fminf(1.f, fmaxf(-1.f, ph));
  out[(size_t)row * DMODEL + tid] = ph * mag * scale[tid];
}

__global__ void swiglu_res_kernel(const float* __restrict__ x, const float* __restrict__ g,
                                  const float* __restrict__ u, float* __restrict__ out,
                                  int n) {
  const int i = blockIdx.x * blockDim.x + threadIdx.x;
  if (i >= n) return;
  const float gv = g[i];
  out[i] = x[i] + gv * sigf(gv) * u[i];
}

// ---------------------------------------------------------------------------
// softmax over 256 experts + top2
// ---------------------------------------------------------------------------
__global__ void top2_kernel(const float* __restrict__ logits, int* __restrict__ idx,
                            float* __restrict__ wts) {
  const int row = blockIdx.x, e = threadIdx.x;
  __shared__ float sv[NEXP];
  __shared__ int si[NEXP];
  const float v = logits[(size_t)row * NEXP + e];
  sv[e] = v;
  __syncthreads();
  for (int s = 128; s > 0; s >>= 1) {
    if (e < s) sv[e] = fmaxf(sv[e], sv[e + s]);
    __syncthreads();
  }
  const float mx = sv[0];
  __syncthreads();
  const float ex = __expf(v - mx);
  sv[e] = ex;
  __syncthreads();
  for (int s = 128; s > 0; s >>= 1) {
    if (e < s) sv[e] += sv[e + s];
    __syncthreads();
  }
  const float p = ex / sv[0];
  __syncthreads();
  sv[e] = p; si[e] = e;
  __syncthreads();
  for (int s = 128; s > 0; s >>= 1) {
    if (e < s && sv[e + s] > sv[e]) { sv[e] = sv[e + s]; si[e] = si[e + s]; }
    __syncthreads();
  }
  const int i0 = si[0];
  const float p0 = sv[0];
  __syncthreads();
  sv[e] = (e == i0) ? -1.f : p; si[e] = e;
  __syncthreads();
  for (int s = 128; s > 0; s >>= 1) {
    if (e < s && sv[e + s] > sv[e]) { sv[e] = sv[e + s]; si[e] = si[e + s]; }
    __syncthreads();
  }
  if (e == 0) {
    const int i1 = si[0];
    const float p1 = sv[0];
    const float tw = p0 + p1 + EPSF;
    idx[row * 2] = i0; idx[row * 2 + 1] = i1;
    wts[row * 2] = p0 / tw; wts[row * 2 + 1] = p1 / tw;
  }
}

// ---------------------------------------------------------------------------
// Expert routing: histogram -> scan -> scatter (4096 entries total)
// ---------------------------------------------------------------------------
__global__ void zero_counts_kernel(int* __restrict__ counts) {
  counts[threadIdx.x] = 0;
}
__global__ void hist_kernel(const int* __restrict__ idx, int* __restrict__ counts) {
  const int i = blockIdx.x * 256 + threadIdx.x;
  if (i >= 2 * NTOK) return;
  atomicAdd(&counts[idx[i]], 1);
}
__global__ void scan_kernel(const int* __restrict__ counts, int* __restrict__ offsets,
                            int* __restrict__ cursor) {
  __shared__ int s[NEXP];
  const int e = threadIdx.x;
  const int c = counts[e];
  s[e] = c;
  __syncthreads();
  for (int d = 1; d < NEXP; d <<= 1) {
    const int v = (e >= d) ? s[e - d] : 0;
    __syncthreads();
    s[e] += v;
    __syncthreads();
  }
  const int excl = s[e] - c;
  offsets[e] = excl;
  cursor[e] = excl;
  if (e == NEXP - 1) offsets[NEXP] = s[e];
}
__global__ void scatter_kernel(const int* __restrict__ idx, int* __restrict__ cursor,
                               int* __restrict__ list) {
  const int i = blockIdx.x * 256 + threadIdx.x;
  if (i >= 2 * NTOK) return;
  const int e = idx[i];
  const int pos = atomicAdd(&cursor[e], 1);
  const int token = i >> 1, slot = i & 1;
  list[pos] = (token << 1) | slot;   // -1 reserved for "empty row"
}

// ---------------------------------------------------------------------------
// Grouped expert GEMM: one block per expert, WMMA over <=16-token tiles.
// E[token | expert] rows are independent -> deterministic despite list order.
// ---------------------------------------------------------------------------
__global__ void expert_wmma_kernel(const float* __restrict__ h,
                                   const _Float16* __restrict__ expw16,
                                   const int* __restrict__ offsets,
                                   const int* __restrict__ list,
                                   float* __restrict__ eo0, float* __restrict__ eo1) {
  const int e = blockIdx.x;
  const int beg = offsets[e];
  const int nt = offsets[e + 1] - beg;
  if (nt == 0) return;

  __shared__ __align__(32) _Float16 As[16 * DMODEL];
  __shared__ int toks[16];
  const int tid = threadIdx.x, lane = tid & 31, wave = tid >> 5;
  const int m = lane & 15, half = lane >> 4;
  const int col0 = wave << 4;
  const _Float16* wcol = expw16 + ((size_t)e << 14) + (size_t)(col0 + m) * DMODEL;

  for (int g = 0; g < nt; g += 16) {
    const int cnt = min(16, nt - g);
    if (tid < 16) toks[tid] = (tid < cnt) ? list[beg + g + tid] : -1;
    __syncthreads();
#pragma unroll
    for (int i = tid; i < 16 * DMODEL; i += 256) {
      const int r = i >> 7, k = i & 127;
      const int te = toks[r];
      As[i] = (te >= 0) ? (_Float16)h[(size_t)(te >> 1) * DMODEL + k] : (_Float16)0.f;
    }
    __syncthreads();

    v8f acc = {0.f, 0.f, 0.f, 0.f, 0.f, 0.f, 0.f, 0.f};
#pragma unroll
    for (int k0 = 0; k0 < DMODEL; k0 += 32) {
      const v8h lo = *(const v8h*)(&As[m * DMODEL + k0 + 16 * half]);
      const v8h hi = *(const v8h*)(&As[m * DMODEL + k0 + 16 * half + 8]);
      v16h a;
#pragma unroll
      for (int i = 0; i < 8; ++i) { a[i] = lo[i]; a[8 + i] = hi[i]; }
      const v16h b = *(const v16h*)(wcol + k0 + 16 * half);
      acc = __builtin_amdgcn_wmma_f32_16x16x32_f16(false, a, false, b, (short)0, acc,
                                                   false, false);
    }
#pragma unroll
    for (int r = 0; r < 8; ++r) {
      const int s = r + 8 * half;
      const int te = toks[s];
      if (te >= 0) {
        float* dst = (te & 1) ? eo1 : eo0;
        dst[(size_t)(te >> 1) * DMODEL + col0 + m] = acc[r];
      }
    }
    __syncthreads();
  }
}

// wavg = w0*eo0 + w1*eo1
__global__ void wavg_kernel(const float* __restrict__ eo0, const float* __restrict__ eo1,
                            const float* __restrict__ wts, float* __restrict__ wavg) {
  const int i = blockIdx.x * 256 + threadIdx.x;
  if (i >= NTOK * DMODEL) return;
  const int row = i >> 7;
  wavg[i] = wts[row * 2] * eo0[i] + wts[row * 2 + 1] * eo1[i];
}

// moe epilogue: swiglu(wavg), variance, consensus, residual
__global__ void moe_final_kernel(const float* __restrict__ x1, const float* __restrict__ g,
                                 const float* __restrict__ u, const float* __restrict__ eo0,
                                 const float* __restrict__ eo1, const float* __restrict__ wts,
                                 float* __restrict__ out_x, float* __restrict__ out_c) {
  const int row = blockIdx.x, c = threadIdx.x;
  const size_t i = (size_t)row * DMODEL + c;
  const float gv = g[i];
  const float ws = gv * sigf(gv) * u[i];
  const float w0 = wts[row * 2], w1 = wts[row * 2 + 1];
  const float d0 = eo0[i] - ws, d1 = eo1[i] - ws;
  out_x[i] = x1[i] + ws;
  __shared__ float red[DMODEL];
  red[c] = w0 * d0 * d0 + w1 * d1 * d1;
  __syncthreads();
  for (int s = 64; s > 0; s >>= 1) {
    if (c < s) red[c] += red[c + s];
    __syncthreads();
  }
  if (c == 0) out_c[row] = __expf(-red[0] / (float)DMODEL);
}

// ---------------------------------------------------------------------------
extern "C" void kernel_launch(void* const* d_in, const int* in_sizes, int n_in,
                              void* d_out, int out_size, void* d_ws, size_t ws_size,
                              hipStream_t stream) {
  const float* x      = (const float*)d_in[0];
  const float* g1     = (const float*)d_in[1];
  const float* qW     = (const float*)d_in[2];
  const float* qb     = (const float*)d_in[3];
  const float* kdW    = (const float*)d_in[4];
  const float* kdb    = (const float*)d_in[5];
  const float* kuW    = (const float*)d_in[6];
  const float* kub    = (const float*)d_in[7];
  const float* oW     = (const float*)d_in[8];
  const float* ob     = (const float*)d_in[9];
  const float* rot    = (const float*)d_in[10];
  const float* tscale = (const float*)d_in[11];
  const float* s1W    = (const float*)d_in[12];
  const float* s1b    = (const float*)d_in[13];
  const float* s2W    = (const float*)d_in[14];
  const float* s2b    = (const float*)d_in[15];
  const float* g2     = (const float*)d_in[16];
  const float* gateW  = (const float*)d_in[17];
  const float* gateb  = (const float*)d_in[18];
  const float* expW   = (const float*)d_in[19];
  const float* m1W    = (const float*)d_in[20];
  const float* m1b    = (const float*)d_in[21];
  const float* m2W    = (const float*)d_in[22];
  const float* m2b    = (const float*)d_in[23];

  float* out_x = (float*)d_out;
  float* out_c = (float*)d_out + (size_t)NTOK * DMODEL;

  const size_t ND = (size_t)NTOK * DMODEL;
  float* ws = (float*)d_ws;
  float* xn     = ws;                    ws += ND;
  float* qbuf   = ws;                    ws += ND;
  float* kd     = ws;                    ws += (size_t)NTOK * RLORA;
  float* kvbuf  = ws;                    ws += (size_t)NTOK * 2 * DMODEL;
  float* attn   = ws;                    ws += ND;
  float* ao     = ws;                    ws += ND;
  float* yrot   = ws;                    ws += ND;
  float* tq     = ws;                    ws += ND;
  float* gbuf   = ws;                    ws += ND;
  float* ubuf   = ws;                    ws += ND;
  float* x1     = ws;                    ws += ND;
  float* hbuf   = ws;                    ws += ND;
  float* logits = ws;                    ws += (size_t)NTOK * NEXP;
  float* eo0    = ws;                    ws += ND;
  float* eo1    = ws;                    ws += ND;
  float* wavg   = ws;                    ws += ND;
  float* wts    = ws;                    ws += (size_t)NTOK * 2;
  int*   idx    = (int*)ws;              ws += (size_t)NTOK * 2;
  int*   counts = (int*)ws;              ws += NEXP;          // reuse float slots as ints
  int*   offs   = (int*)ws;              ws += NEXP + 2;
  int*   cursor = (int*)ws;              ws += NEXP;
  int*   list   = (int*)ws;              ws += 2 * NTOK;

  // f16 transposed weight area (only 8B aligned: all offsets above are even floats)
  _Float16* h16 = (_Float16*)ws;
  _Float16* qW16   = h16;  h16 += DMODEL * DMODEL;
  _Float16* kdW16  = h16;  h16 += DMODEL * RLORA;
  _Float16* kuW16  = h16;  h16 += RLORA * 2 * DMODEL;
  _Float16* oW16   = h16;  h16 += DMODEL * DMODEL;
  _Float16* rot16  = h16;  h16 += DMODEL * DMODEL;
  _Float16* s1W16  = h16;  h16 += DMODEL * DMODEL;
  _Float16* s2W16  = h16;  h16 += DMODEL * DMODEL;
  _Float16* gate16 = h16;  h16 += DMODEL * NEXP;
  _Float16* m1W16  = h16;  h16 += DMODEL * DMODEL;
  _Float16* m2W16  = h16;  h16 += DMODEL * DMODEL;
  _Float16* expw16 = h16;  h16 += (size_t)NEXP * DMODEL * DMODEL;

  const dim3 blk256(256), blk128(128);
  const int RT = NTOK / 16;

  auto cvt = [&](const float* W, _Float16* Wt, int K, int M) {
    const int n = K * M;
    wt_prep_kernel<<<(n + 255) / 256, blk256, 0, stream>>>(W, Wt, K, M);
  };
  cvt(qW, qW16, DMODEL, DMODEL);
  cvt(kdW, kdW16, DMODEL, RLORA);
  cvt(kuW, kuW16, RLORA, 2 * DMODEL);
  cvt(oW, oW16, DMODEL, DMODEL);
  cvt(rot, rot16, DMODEL, DMODEL);
  cvt(s1W, s1W16, DMODEL, DMODEL);
  cvt(s2W, s2W16, DMODEL, DMODEL);
  cvt(gateW, gate16, DMODEL, NEXP);
  cvt(m1W, m1W16, DMODEL, DMODEL);
  cvt(m2W, m2W16, DMODEL, DMODEL);
  {
    const size_t n = (size_t)NEXP * DMODEL * DMODEL;
    expw_prep_kernel<<<(int)((n + 255) / 256), blk256, 0, stream>>>(expW, expw16);
  }

  // --- attention path ---
  rmsnorm_kernel<<<NTOK, blk128, 0, stream>>>(x, g1, xn);
  gemm_wmma<DMODEL, ACT_ELU_ALL, true><<<dim3(RT, 1), blk256, 0, stream>>>(xn, qW16, qb, qbuf, DMODEL);
  gemm_wmma<DMODEL, ACT_NONE, true><<<dim3(RT, 1), blk256, 0, stream>>>(xn, kdW16, kdb, kd, RLORA);
  gemm_wmma<RLORA, ACT_ELU_KHALF, true><<<dim3(RT, 2), blk256, 0, stream>>>(kd, kuW16, kub, kvbuf, 2 * DMODEL);
  attn_kernel<<<4 * NH, blk256, 0, stream>>>(qbuf, kvbuf, attn);
  gemm_wmma<DMODEL, ACT_NONE, true><<<dim3(RT, 1), blk256, 0, stream>>>(attn, oW16, ob, ao, DMODEL);
  gemm_wmma<DMODEL, ACT_NONE, false><<<dim3(RT, 1), blk256, 0, stream>>>(ao, rot16, nullptr, yrot, DMODEL);
  tquant_kernel<<<NTOK, blk128, 0, stream>>>(yrot, tscale, tq);
  gemm_wmma<DMODEL, ACT_NONE, true><<<dim3(RT, 1), blk256, 0, stream>>>(tq, s1W16, s1b, gbuf, DMODEL);
  gemm_wmma<DMODEL, ACT_NONE, true><<<dim3(RT, 1), blk256, 0, stream>>>(tq, s2W16, s2b, ubuf, DMODEL);
  swiglu_res_kernel<<<(int)(ND / 256), blk256, 0, stream>>>(x, gbuf, ubuf, x1, (int)ND);

  // --- MoE path: top-2 sparse (identical through the mask), grouped-by-expert WMMA ---
  rmsnorm_kernel<<<NTOK, blk128, 0, stream>>>(x1, g2, hbuf);
  gemm_wmma<DMODEL, ACT_NONE, true><<<dim3(RT, 2), blk256, 0, stream>>>(hbuf, gate16, gateb, logits, NEXP);
  top2_kernel<<<NTOK, blk256, 0, stream>>>(logits, idx, wts);
  zero_counts_kernel<<<1, NEXP, 0, stream>>>(counts);
  hist_kernel<<<(2 * NTOK + 255) / 256, blk256, 0, stream>>>(idx, counts);
  scan_kernel<<<1, NEXP, 0, stream>>>(counts, offs, cursor);
  scatter_kernel<<<(2 * NTOK + 255) / 256, blk256, 0, stream>>>(idx, cursor, list);
  expert_wmma_kernel<<<NEXP, blk256, 0, stream>>>(hbuf, expw16, offs, list, eo0, eo1);
  wavg_kernel<<<(int)(ND / 256), blk256, 0, stream>>>(eo0, eo1, wts, wavg);
  gemm_wmma<DMODEL, ACT_NONE, true><<<dim3(RT, 1), blk256, 0, stream>>>(wavg, m1W16, m1b, gbuf, DMODEL);
  gemm_wmma<DMODEL, ACT_NONE, true><<<dim3(RT, 1), blk256, 0, stream>>>(wavg, m2W16, m2b, ubuf, DMODEL);
  moe_final_kernel<<<NTOK, blk128, 0, stream>>>(x1, gbuf, ubuf, eo0, eo1, wts, out_x, out_c);
}
